// HybridCrissCrossTransformer_9981503996363
// MI455X (gfx1250) — matrix operand order repaired and checked by reading, not compiled, not run on hardware
//
#include <hip/hip_runtime.h>
#include <hip/hip_bf16.h>

// HybridCrissCrossTransformer for MI455X (gfx1250, wave32, WMMA + TDM).
// All matmuls via v_wmma_f32_16x16x32_f16; GEMM B-tiles staged into LDS by the
// Tensor Data Mover (tensor_load_to_lds); softmax/LayerNorm in fp32.

typedef _Float16 f16;
typedef __attribute__((ext_vector_type(16))) _Float16     v16h;
typedef __attribute__((ext_vector_type(8)))  float        v8f;
typedef __attribute__((ext_vector_type(4)))  unsigned int v4u;
typedef __attribute__((ext_vector_type(8)))  unsigned int v8u;

#define B_    8
#define C_    512
#define HW_   4096
#define NH_   8
#define HD_   64
#define FF_   2048
#define MTOT_ 32768   // B*HW

// ---------------- WMMA helpers (A/B k order: contiguous, k = 16*(lane/16) + s) ----------------

__device__ __forceinline__ v8f wmma_f16(v16h a, v16h b, v8f c) {
  return __builtin_amdgcn_wmma_f32_16x16x32_f16(
      /*neg_a=*/false, a, /*neg_b=*/false, b,
      /*c_mod=*/(short)0, c, /*reuse_a=*/false, /*reuse_b=*/false);
}

// A-matrix 16x32 f16: lanes 0-15 -> M=lane, K=0..15;
// lanes 16-31 -> M=lane-16, K=16..31. As is row-major [M][lda].
__device__ __forceinline__ v16h load_a_frag(const f16* As, int lda, int mbase) {
  const int lane = threadIdx.x & 31;
  const int r = lane & 15, hi = lane >> 4;
  const f16* p = As + (mbase + r) * lda;
  v16h a;
#pragma unroll
  for (int j = 0; j < 8; ++j) {
    const int k0 = hi * 16 + 2 * j;
    a[2 * j]     = p[k0];
    a[2 * j + 1] = p[k0 + 1];
  }
  return a;
}

// B-matrix 32x16 f16: lanes 0-15 -> N=lane, K=0..15; lanes 16-31 -> K=16..31.
// Bs is row-major [K][ldb] (K chunk of 32 at Bs).
__device__ __forceinline__ v16h load_b_frag(const f16* Bs, int ldb, int nbase) {
  const int lane = threadIdx.x & 31;
  const int c = (lane & 15) + nbase, hi = lane >> 4;
  v16h b;
#pragma unroll
  for (int j = 0; j < 8; ++j) {
    const int k = hi * 16 + 2 * j;
    b[2 * j]     = Bs[k * ldb + c];
    b[2 * j + 1] = Bs[(k + 1) * ldb + c];
  }
  return b;
}

// B fragment of Ks^T where Ks is row-major [N][ldk] (i.e. B[k][n] = Ks[n][k]).
__device__ __forceinline__ v16h load_bT_frag(const f16* Ks, int ldk, int nbase) {
  const int lane = threadIdx.x & 31;
  const int c = (lane & 15) + nbase, hi = lane >> 4;
  const f16* p = Ks + c * ldk;
  v16h b;
#pragma unroll
  for (int j = 0; j < 8; ++j) {
    const int k = hi * 16 + 2 * j;
    b[2 * j]     = p[k];
    b[2 * j + 1] = p[k + 1];
  }
  return b;
}

// ---------------- TDM: 2D tile global -> LDS (ISA 8.3/8.4 D#) ----------------
// Loads `nrows` rows of `row_bytes` bytes (global row stride `stride_bytes`)
// starting at `gaddr` into LDS byte offset `lds_addr`, densely packed.
// data_size = 1 byte; 2-group descriptor (2D tensor); no multicast/pad/iterate.

__device__ __forceinline__ void tdm_load_2d(unsigned lds_addr, const void* gaddr,
                                            unsigned row_bytes, unsigned nrows,
                                            unsigned stride_bytes) {
  const unsigned long long ga = (unsigned long long)gaddr;
  const unsigned ga_lo = __builtin_amdgcn_readfirstlane((unsigned)ga);
  const unsigned ga_hi = __builtin_amdgcn_readfirstlane((unsigned)(ga >> 32));
  const unsigned la    = __builtin_amdgcn_readfirstlane(lds_addr);
  const unsigned st    = __builtin_amdgcn_readfirstlane(stride_bytes);

  v4u g0;
  g0[0] = 1u;                                        // count=1, user descriptor
  g0[1] = la;                                        // lds_addr [63:32]
  g0[2] = ga_lo;                                     // global_addr [95:64]
  g0[3] = (ga_hi & 0x01FFFFFFu) | 0x80000000u;       // ga[56:32] | type=2

  v8u g1;
  g1[0] = 0u;                                        // mask=0, data_size=1B
  g1[1] = (row_bytes & 0xFFFFu) << 16;               // tensor_dim0 [15:0]
  g1[2] = (row_bytes >> 16) | ((nrows & 0xFFFFu) << 16);  // tdim0 hi | tdim1 lo
  g1[3] = (nrows >> 16) | ((row_bytes & 0xFFFFu) << 16);  // tdim1 hi | tile_dim0
  g1[4] = nrows & 0xFFFFu;                           // tile_dim1 (tile_dim2=0)
  g1[5] = st;                                        // tensor_dim0_stride lo32
  g1[6] = 0u;                                        // stride hi | tdim1_stride lo
  g1[7] = 0u;

  asm volatile("tensor_load_to_lds %0, %1" :: "s"(g0), "s"(g1) : "memory");
}

__device__ __forceinline__ unsigned lds_offset_of(const void* p) {
  // generic LDS pointer: low 32 bits are the wave-relative LDS byte offset
  return (unsigned)(unsigned long long)p;
}

// ---------------- kernel 0: fp32 -> f16 weight conversion ----------------

__global__ __launch_bounds__(256) void cvt_f16_kernel(const float* __restrict__ src,
                                                      f16* __restrict__ dst, int n) {
  int i = blockIdx.x * 256 + threadIdx.x;
  if (i < n) dst[i] = (f16)src[i];
}

// ------- kernel 1: x[B,C,H,W] -> resid fp32 [B,HW,C] + xh f16 [B,HW,C] ------

__global__ __launch_bounds__(256) void prep_kernel(const float* __restrict__ x,
                                                   float* __restrict__ resid,
                                                   f16* __restrict__ xh) {
  __shared__ float tile[32][33];
  const int b = blockIdx.z;
  const int pbase = blockIdx.x * 32;  // HW tile
  const int cbase = blockIdx.y * 32;  // C tile
  const int tx = threadIdx.x & 31, ty = threadIdx.x >> 5;
  const float* xb = x + ((size_t)b * C_ + cbase) * HW_ + pbase;
#pragma unroll
  for (int j = 0; j < 4; ++j) {
    const int c = ty + j * 8;
    tile[c][tx] = xb[(size_t)c * HW_ + tx];
  }
  __syncthreads();
#pragma unroll
  for (int j = 0; j < 4; ++j) {
    const int p = ty + j * 8;
    const float v = tile[tx][p];
    const size_t o = ((size_t)b * HW_ + pbase + p) * C_ + cbase + tx;
    resid[o] = v;
    xh[o] = (f16)v;
  }
}

// ---------------- kernel 2/5: generic 128x128 WMMA GEMM ----------------
// B-tile (32 x 128 f16) staged into LDS by the TDM each K step.
// EPI==1: ReLU epilogue, row-major f16 out (FFN1)
// EPI==2: QKV epilogue, scatter into [B,NH,HW,HD] f16

template <int EPI>
__global__ __launch_bounds__(256) void gemm_f16_kernel(
    const f16* __restrict__ A, const f16* __restrict__ Bw,
    const float* __restrict__ bias, f16* __restrict__ out, int N, int K) {
  __shared__ f16 As[128 * 32];
  __shared__ f16 Bs[32 * 128];
  const int tid = threadIdx.x;
  const int lane = tid & 31, wave = tid >> 5;
  const int wm = wave >> 2, wn = wave & 3;  // 2 x 4 wave grid
  const long rowbase = (long)blockIdx.x * 128;
  const int colbase = blockIdx.y * 128;
  const unsigned bs_lds = lds_offset_of(Bs);

  const v8f zero = {0.f, 0.f, 0.f, 0.f, 0.f, 0.f, 0.f, 0.f};
  v8f acc[4][2];
#pragma unroll
  for (int mt = 0; mt < 4; ++mt)
#pragma unroll
    for (int nt = 0; nt < 2; ++nt) acc[mt][nt] = zero;

  for (int k0 = 0; k0 < K; k0 += 32) {
    // A tile: 128 rows x 32 halves via VGPRs
#pragma unroll
    for (int it = 0; it < 2; ++it) {
      const int idx = tid + it * 256;
      const int r = idx >> 2, g = idx & 3;  // 128 rows x 4 groups of 8 halves
      *(float4*)(As + r * 32 + g * 8) =
          *(const float4*)(A + (rowbase + r) * (long)K + k0 + g * 8);
    }
    // B tile: 32 rows x 256 bytes, row stride N*2 bytes — one TDM DMA (wave 0)
    if (wave == 0) {
      tdm_load_2d(bs_lds, Bw + (long)k0 * N + colbase,
                  /*row_bytes=*/256u, /*nrows=*/32u,
                  /*stride_bytes=*/(unsigned)N * 2u);
      __builtin_amdgcn_s_wait_tensorcnt(0);
    }
    if (k0 + 32 < K)  // gfx1250 global_prefetch of next A chunk
      __builtin_prefetch(A + (rowbase + (tid >> 2)) * (long)K + k0 + 32, 0, 1);
    __syncthreads();
    v16h af[4], bf[2];
#pragma unroll
    for (int mt = 0; mt < 4; ++mt) af[mt] = load_a_frag(As, 32, wm * 64 + mt * 16);
#pragma unroll
    for (int nt = 0; nt < 2; ++nt) bf[nt] = load_b_frag(Bs, 128, wn * 32 + nt * 16);
#pragma unroll
    for (int mt = 0; mt < 4; ++mt)
#pragma unroll
      for (int nt = 0; nt < 2; ++nt)
        acc[mt][nt] = wmma_f16(af[mt], bf[nt], acc[mt][nt]);
    __syncthreads();
  }

  // C/D layout: VGPR i -> row (hi*8 + i), col = lane&15
  const int c0 = lane & 15, hi = lane >> 4;
#pragma unroll
  for (int mt = 0; mt < 4; ++mt)
#pragma unroll
    for (int nt = 0; nt < 2; ++nt)
#pragma unroll
      for (int i = 0; i < 8; ++i) {
        const long row = rowbase + wm * 64 + mt * 16 + hi * 8 + i;
        const int col = colbase + wn * 32 + nt * 16 + c0;
        float v = acc[mt][nt][i] + bias[col];
        if (EPI == 1) v = fmaxf(v, 0.0f);
        if (EPI == 2) {
          const int n = col >> 6, d = col & 63;
          const long bb = row >> 12, pp = row & 4095;
          out[(((bb * NH_ + n) * (long)HW_) + pp) * HD_ + d] = (f16)v;
        } else {
          out[row * (long)N + col] = (f16)v;
        }
      }
}

// ---------------- kernel 3: criss-cross attention ----------------
// grid.x in [0,128): <64 -> row attention (fixed w), >=64 -> col attention
// (fixed h). grid.y = b*NH + n. 4 waves, WMMA over LDS-resident 64x64 tiles.

__global__ __launch_bounds__(128) void attn_kernel(const f16* __restrict__ qh,
                                                   const f16* __restrict__ kh,
                                                   const f16* __restrict__ vh,
                                                   float* __restrict__ orow,
                                                   float* __restrict__ ocol) {
  __shared__ f16 Qs[64 * 64];  // reused as P after softmax
  __shared__ f16 Ks[64 * 64];
  __shared__ f16 Vs[64 * 64];
  __shared__ float Ss[64 * 64];
  const int tid = threadIdx.x, lane = tid & 31, wave = tid >> 5;
  const int line = blockIdx.x;
  const int bn = blockIdx.y;
  const bool isRow = line < 64;
  const int fixed = isRow ? line : line - 64;  // w (row-attn) or h (col-attn)
  const size_t base = (size_t)bn * HW_ * HD_;

  // gather the 64-long sequence (stride W for row-attn, contiguous for col-attn)
  for (int idx = tid; idx < 512; idx += 128) {
    const int s = idx >> 3, g = idx & 7;
    const int p = isRow ? (s * 64 + fixed) : (fixed * 64 + s);
    const size_t off = base + (size_t)p * HD_ + g * 8;
    *(float4*)(Qs + s * 64 + g * 8) = *(const float4*)(qh + off);
    *(float4*)(Ks + s * 64 + g * 8) = *(const float4*)(kh + off);
    *(float4*)(Vs + s * 64 + g * 8) = *(const float4*)(vh + off);
  }
  __syncthreads();

  const v8f zero = {0.f, 0.f, 0.f, 0.f, 0.f, 0.f, 0.f, 0.f};
  const int mbase = wave * 16;
  const int c0 = lane & 15, hi = lane >> 4;

  // S = Q K^T  (64x64, reduce over d=64)
  {
    v8f acc[4];
#pragma unroll
    for (int nt = 0; nt < 4; ++nt) acc[nt] = zero;
#pragma unroll
    for (int kk = 0; kk < 64; kk += 32) {
      const v16h a = load_a_frag(Qs + kk, 64, mbase);
#pragma unroll
      for (int nt = 0; nt < 4; ++nt) {
        const v16h b = load_bT_frag(Ks + kk, 64, nt * 16);
        acc[nt] = wmma_f16(a, b, acc[nt]);
      }
    }
#pragma unroll
    for (int nt = 0; nt < 4; ++nt)
#pragma unroll
      for (int i = 0; i < 8; ++i)
        Ss[(mbase + hi * 8 + i) * 64 + nt * 16 + c0] = acc[nt][i];
  }
  __syncthreads();

  // softmax over the 64 keys; write P (f16) into Qs
  for (int rr = 0; rr < 16; ++rr) {
    const int r = mbase + rr;
    const float v0 = Ss[r * 64 + lane];
    const float v1 = Ss[r * 64 + 32 + lane];
    float m = fmaxf(v0, v1);
#pragma unroll
    for (int o = 16; o > 0; o >>= 1) m = fmaxf(m, __shfl_xor(m, o, 32));
    const float e0 = __expf(v0 - m), e1 = __expf(v1 - m);
    float s = e0 + e1;
#pragma unroll
    for (int o = 16; o > 0; o >>= 1) s += __shfl_xor(s, o, 32);
    const float inv = 1.0f / s;
    Qs[r * 64 + lane] = (f16)(e0 * inv);
    Qs[r * 64 + 32 + lane] = (f16)(e1 * inv);
  }
  __syncthreads();

  // O = P V  (64x64, reduce over seq=64)
  {
    v8f acc[4];
#pragma unroll
    for (int nt = 0; nt < 4; ++nt) acc[nt] = zero;
#pragma unroll
    for (int kk = 0; kk < 64; kk += 32) {
      const v16h a = load_a_frag(Qs + kk, 64, mbase);
#pragma unroll
      for (int nt = 0; nt < 4; ++nt) {
        const v16h b = load_b_frag(Vs + kk * 64, 64, nt * 16);
        acc[nt] = wmma_f16(a, b, acc[nt]);
      }
    }
    float* outp = isRow ? orow : ocol;
#pragma unroll
    for (int nt = 0; nt < 4; ++nt)
#pragma unroll
      for (int i = 0; i < 8; ++i) {
        const int r = mbase + hi * 8 + i;  // query index along the line
        const int p = isRow ? (r * 64 + fixed) : (fixed * 64 + r);
        outp[base + (size_t)p * HD_ + nt * 16 + c0] = acc[nt][i];
      }
  }
}

// ---------------- kernel 4: y = LN(resid + gamma*(orow+ocol)) ----------------

__global__ __launch_bounds__(256) void ln1_kernel(
    const float* __restrict__ resid, const float* __restrict__ orow,
    const float* __restrict__ ocol, const float* __restrict__ g1,
    const float* __restrict__ bb1, const float* __restrict__ gamma,
    float* __restrict__ y, f16* __restrict__ yh) {
  const int p = blockIdx.x;  // 0..32767
  const int tid = threadIdx.x;
  const int b = p >> 12, pp = p & 4095;
  const float gm = gamma[0];
  float v[2];
#pragma unroll
  for (int j = 0; j < 2; ++j) {
    const int c = tid + j * 256;
    const int n = c >> 6, d = c & 63;
    const size_t ai = (((size_t)(b * NH_ + n)) * HW_ + pp) * HD_ + d;
    v[j] = resid[(size_t)p * C_ + c] + gm * (orow[ai] + ocol[ai]);
  }
  float s = v[0] + v[1];
  float ss = v[0] * v[0] + v[1] * v[1];
#pragma unroll
  for (int o = 16; o > 0; o >>= 1) {
    s += __shfl_xor(s, o, 32);
    ss += __shfl_xor(ss, o, 32);
  }
  __shared__ float ps[8], pss[8];
  const int wave = tid >> 5, lane = tid & 31;
  if (lane == 0) { ps[wave] = s; pss[wave] = ss; }
  __syncthreads();
  if (tid == 0) {
    float S = 0.f, SS = 0.f;
#pragma unroll
    for (int i = 0; i < 8; ++i) { S += ps[i]; SS += pss[i]; }
    ps[0] = S; pss[0] = SS;
  }
  __syncthreads();
  const float mu = ps[0] * (1.0f / 512.0f);
  const float var = pss[0] * (1.0f / 512.0f) - mu * mu;
  const float rstd = rsqrtf(var + 1e-5f);
#pragma unroll
  for (int j = 0; j < 2; ++j) {
    const int c = tid + j * 256;
    const float o = (v[j] - mu) * rstd * g1[c] + bb1[c];
    y[(size_t)p * C_ + c] = o;
    yh[(size_t)p * C_ + c] = (f16)o;
  }
}

// -------- kernel 6: ff = h1@W2 + b2; out = LN(y+ff) transposed to [B,C,H,W] -----
// Tile 32 rows x 512 cols -> whole LN row inside one block (fused).
// W2 tile (32 x 512 f16 = 32 KB) staged by the TDM each K step.

__global__ __launch_bounds__(256) void ffn2_ln2_kernel(
    const f16* __restrict__ h1, const f16* __restrict__ W2h,
    const float* __restrict__ b2, const float* __restrict__ y,
    const float* __restrict__ g2, const float* __restrict__ bb2,
    float* __restrict__ out) {
  __shared__ f16 As[32 * 32];
  __shared__ f16 Bs[32 * 512];
  __shared__ float red[32][4][2];
  const int tid = threadIdx.x, lane = tid & 31, wave = tid >> 5;
  const int wm = wave >> 2, wn = wave & 3;  // 2 x 4 wave grid, wave tile 16x128
  const long rowbase = (long)blockIdx.x * 32;
  const unsigned bs_lds = lds_offset_of(Bs);

  const v8f zero = {0.f, 0.f, 0.f, 0.f, 0.f, 0.f, 0.f, 0.f};
  v8f acc[8];
#pragma unroll
  for (int nt = 0; nt < 8; ++nt) acc[nt] = zero;

  for (int k0 = 0; k0 < FF_; k0 += 32) {
    if (tid < 128) {
      const int r = tid >> 2, g = tid & 3;  // 32 rows x 4 groups
      *(float4*)(As + r * 32 + g * 8) =
          *(const float4*)(h1 + (rowbase + r) * (long)FF_ + k0 + g * 8);
    }
    if (wave == 0) {  // 32 rows x 1024 bytes, contiguous (stride == width)
      tdm_load_2d(bs_lds, W2h + (long)k0 * C_,
                  /*row_bytes=*/1024u, /*nrows=*/32u, /*stride_bytes=*/1024u);
      __builtin_amdgcn_s_wait_tensorcnt(0);
    }
    __syncthreads();
    const v16h a = load_a_frag(As, 32, wm * 16);
#pragma unroll
    for (int nt = 0; nt < 8; ++nt) {
      const v16h b = load_b_frag(Bs, 512, wn * 128 + nt * 16);
      acc[nt] = wmma_f16(a, b, acc[nt]);
    }
    __syncthreads();
  }

  const int c0 = lane & 15, hi = lane >> 4;
  // v = y + ff + b2
#pragma unroll
  for (int nt = 0; nt < 8; ++nt)
#pragma unroll
    for (int i = 0; i < 8; ++i) {
      const long row = rowbase + wm * 16 + hi * 8 + i;
      const int col = wn * 128 + nt * 16 + c0;
      acc[nt][i] += b2[col] + y[row * (long)C_ + col];
    }
  // per-row partial sums over this wave's 128 cols (16 lanes of one half x 8 tiles)
#pragma unroll
  for (int i = 0; i < 8; ++i) {
    float s = 0.f, ss = 0.f;
#pragma unroll
    for (int nt = 0; nt < 8; ++nt) {
      const float t = acc[nt][i];
      s += t;
      ss += t * t;
    }
#pragma unroll
    for (int o = 8; o > 0; o >>= 1) {  // reduce within 16-lane half
      s += __shfl_xor(s, o, 32);
      ss += __shfl_xor(ss, o, 32);
    }
    if (c0 == 0) {
      const int rl = wm * 16 + hi * 8 + i;
      red[rl][wn][0] = s;
      red[rl][wn][1] = ss;
    }
  }
  __syncthreads();
  float mu8[8], rs8[8];
#pragma unroll
  for (int i = 0; i < 8; ++i) {
    const int rl = wm * 16 + hi * 8 + i;
    float S = 0.f, SS = 0.f;
#pragma unroll
    for (int w = 0; w < 4; ++w) { S += red[rl][w][0]; SS += red[rl][w][1]; }
    const float mu = S * (1.0f / 512.0f);
    mu8[i] = mu;
    rs8[i] = rsqrtf(SS * (1.0f / 512.0f) - mu * mu + 1e-5f);
  }
  // normalize + scatter to [B,C,H,W]
#pragma unroll
  for (int nt = 0; nt < 8; ++nt)
#pragma unroll
    for (int i = 0; i < 8; ++i) {
      const long row = rowbase + wm * 16 + hi * 8 + i;
      const int col = wn * 128 + nt * 16 + c0;
      const float o = (acc[nt][i] - mu8[i]) * rs8[i] * g2[col] + bb2[col];
      const long bb = row >> 12, pp = row & 4095;
      out[((long)bb * C_ + col) * HW_ + pp] = o;
    }
}

// ---------------- host-side launch ----------------

extern "C" void kernel_launch(void* const* d_in, const int* in_sizes, int n_in,
                              void* d_out, int out_size, void* d_ws, size_t ws_size,
                              hipStream_t stream) {
  const float* x     = (const float*)d_in[0];
  const float* Wq    = (const float*)d_in[1];
  const float* bq    = (const float*)d_in[2];
  const float* Wk    = (const float*)d_in[3];
  const float* bk    = (const float*)d_in[4];
  const float* Wv    = (const float*)d_in[5];
  const float* bv    = (const float*)d_in[6];
  const float* W1    = (const float*)d_in[7];
  const float* b1    = (const float*)d_in[8];
  const float* W2    = (const float*)d_in[9];
  const float* b2    = (const float*)d_in[10];
  const float* ln1g  = (const float*)d_in[11];
  const float* ln1b  = (const float*)d_in[12];
  const float* ln2g  = (const float*)d_in[13];
  const float* ln2b  = (const float*)d_in[14];
  const float* gamma = (const float*)d_in[15];
  float* out = (float*)d_out;

  char* ws = (char*)d_ws;
  // workspace layout (bytes), ~550 MB total
  float* resid = (float*)(ws + 0);            // 67108864
  f16*   xh    = (f16*)(ws + 67108864);       // 33554432
  f16*   qh    = (f16*)(ws + 100663296);      // 33554432
  f16*   kh    = (f16*)(ws + 134217728);      // 33554432
  f16*   vh    = (f16*)(ws + 167772160);      // 33554432
  float* orow  = (float*)(ws + 201326592);    // 67108864
  float* ocol  = (float*)(ws + 268435456);    // 67108864
  float* y     = (float*)(ws + 335544320);    // 67108864
  f16*   yh    = (f16*)(ws + 402653184);      // 33554432
  f16*   h1    = (f16*)(ws + 436207616);      // 134217728
  f16*   wqh   = (f16*)(ws + 570425344);      // 524288
  f16*   wkh   = (f16*)(ws + 570949632);      // 524288
  f16*   wvh   = (f16*)(ws + 571473920);      // 524288
  f16*   w1h   = (f16*)(ws + 571998208);      // 2097152
  f16*   w2h   = (f16*)(ws + 574095360);      // 2097152

  // 0) weights -> f16
  cvt_f16_kernel<<<(C_ * C_ + 255) / 256, 256, 0, stream>>>(Wq, wqh, C_ * C_);
  cvt_f16_kernel<<<(C_ * C_ + 255) / 256, 256, 0, stream>>>(Wk, wkh, C_ * C_);
  cvt_f16_kernel<<<(C_ * C_ + 255) / 256, 256, 0, stream>>>(Wv, wvh, C_ * C_);
  cvt_f16_kernel<<<(C_ * FF_ + 255) / 256, 256, 0, stream>>>(W1, w1h, C_ * FF_);
  cvt_f16_kernel<<<(C_ * FF_ + 255) / 256, 256, 0, stream>>>(W2, w2h, C_ * FF_);

  // 1) transpose to [B,HW,C] (fp32 residual + f16)
  prep_kernel<<<dim3(HW_ / 32, C_ / 32, B_), 256, 0, stream>>>(x, resid, xh);

  // 2) Q,K,V projections -> per-head f16 layout [B,NH,HW,HD]
  gemm_f16_kernel<2><<<dim3(MTOT_ / 128, C_ / 128), 256, 0, stream>>>(xh, wqh, bq, qh, C_, C_);
  gemm_f16_kernel<2><<<dim3(MTOT_ / 128, C_ / 128), 256, 0, stream>>>(xh, wkh, bk, kh, C_, C_);
  gemm_f16_kernel<2><<<dim3(MTOT_ / 128, C_ / 128), 256, 0, stream>>>(xh, wvh, bv, vh, C_, C_);

  // 3) criss-cross attention (64 row-lines + 64 col-lines per head)
  attn_kernel<<<dim3(128, B_ * NH_), 128, 0, stream>>>(qh, kh, vh, orow, ocol);

  // 4) y = LN1(resid + gamma*(orow+ocol))
  ln1_kernel<<<MTOT_, 256, 0, stream>>>(resid, orow, ocol, ln1g, ln1b, gamma, y, yh);

  // 5) h1 = relu(y@W1 + b1)
  gemm_f16_kernel<1><<<dim3(MTOT_ / 128, FF_ / 128), 256, 0, stream>>>(yh, w1h, b1, h1, FF_, C_);

  // 6) out = LN2(y + h1@W2 + b2), transposed back to [B,C,H,W]
  ffn2_ln2_kernel<<<MTOT_ / 32, 256, 0, stream>>>(h1, w2h, b2, y, ln2g, ln2b, out);
}
